// GnnEncoder_45260365365373
// MI455X (gfx1250) — compile-verified, matrix operand order checked by reading, not run on hardware
//
#include <hip/hip_runtime.h>
#include <hip/hip_bf16.h>
#include <stdint.h>

// ---------------------------------------------------------------------------
// GraphSAGE 3-layer encoder for MI455X (gfx1250, wave32, WMMA).
//  - scatter-mean aggregation via L2-resident f32 atomics
//  - per layer: fused GEMM  out = [mean | h] @ [Wl | Wr]^T + b  using
//    v_wmma_f32_16x16x32_bf16 (bf16 inputs, fp32 accumulate), fused ReLU
//  - LayerNorm as a wave-per-row kernel
// ---------------------------------------------------------------------------

typedef __attribute__((ext_vector_type(16))) __bf16 v16bf;
typedef __attribute__((ext_vector_type(8)))  __bf16 v8bf;
typedef __attribute__((ext_vector_type(8)))  float  v8f;

#define N_NODES 50000
#define N_EDGES 800000
#define KDIM    256          // concatenated [mean | h] reduction dim

// ------------------------------ utility kernels -----------------------------

__global__ void k_zero_f32(float* __restrict__ p, long long n) {
  long long i = (long long)blockIdx.x * blockDim.x + threadIdx.x;
  long long stride = (long long)gridDim.x * blockDim.x;
  for (; i < n; i += stride) p[i] = 0.0f;
}

__global__ void k_count(const int* __restrict__ dst, float* __restrict__ cnt, int E) {
  int e = blockIdx.x * blockDim.x + threadIdx.x;
  if (e >= E) return;
  __hip_atomic_fetch_add(&cnt[dst[e]], 1.0f, __ATOMIC_RELAXED, __HIP_MEMORY_SCOPE_AGENT);
}

__global__ void k_invert(float* __restrict__ cnt, int n) {
  int i = blockIdx.x * blockDim.x + threadIdx.x;
  if (i >= n) return;
  cnt[i] = 1.0f / fmaxf(cnt[i], 1.0f);
}

// One wave (32 lanes) per edge: 128 floats = 4 per lane, L2-resident atomics.
__global__ void k_aggregate(const float* __restrict__ h,
                            const int* __restrict__ src,
                            const int* __restrict__ dst,
                            float* __restrict__ sums, int E) {
  int lane = threadIdx.x & 31;
  int e = (blockIdx.x * blockDim.x + threadIdx.x) >> 5;
  if (e >= E) return;
  int s = src[e], d = dst[e];
  float4 v = ((const float4*)(h + (size_t)s * 128))[lane];
  float* o = sums + (size_t)d * 128 + lane * 4;
  __hip_atomic_fetch_add(o + 0, v.x, __ATOMIC_RELAXED, __HIP_MEMORY_SCOPE_AGENT);
  __hip_atomic_fetch_add(o + 1, v.y, __ATOMIC_RELAXED, __HIP_MEMORY_SCOPE_AGENT);
  __hip_atomic_fetch_add(o + 2, v.z, __ATOMIC_RELAXED, __HIP_MEMORY_SCOPE_AGENT);
  __hip_atomic_fetch_add(o + 3, v.w, __ATOMIC_RELAXED, __HIP_MEMORY_SCOPE_AGENT);
}

// act[n, 0:128)   = bf16(sums[n,:] * inv[n])   (mean)
// act[n, 128:256) = bf16(h[n,:])               (root features)
__global__ void k_build_act(const float* __restrict__ sums,
                            const float* __restrict__ inv,
                            const float* __restrict__ h,
                            __bf16* __restrict__ act, int N) {
  int n = blockIdx.x;
  int c = threadIdx.x;               // 128 threads
  if (n >= N) return;
  float m = sums[(size_t)n * 128 + c] * inv[n];
  act[(size_t)n * KDIM + c]       = (__bf16)m;
  act[(size_t)n * KDIM + 128 + c] = (__bf16)h[(size_t)n * 128 + c];
}

// Wc[o, k] = k<128 ? Wl[o,k] : Wr[o,k-128]   (bf16, row-major [Cout x 256])
__global__ void k_pack_w(const float* __restrict__ Wl, const float* __restrict__ Wr,
                         __bf16* __restrict__ Wc, int Cout) {
  int idx = blockIdx.x * blockDim.x + threadIdx.x;
  if (idx >= Cout * KDIM) return;
  int o = idx >> 8, k = idx & 255;
  float v = (k < 128) ? Wl[o * 128 + k] : Wr[o * 128 + (k - 128)];
  Wc[idx] = (__bf16)v;
}

// ------------------------------- WMMA GEMM ---------------------------------
// One wave per 16-row block; NTILES N-tiles of 16 cols; K=256 in 8 steps of 32.
// Fragment layout per ISA 7.12.2 (16-bit A 16x32): lane L holds row (L&15),
// K-halves (L>>4)*8 .. +7 and +16 .. +23  -> two 16B loads.

__device__ __forceinline__ v16bf load_frag(const __bf16* p) {
  v8bf lo = *(const v8bf*)p;
  v8bf hi = *(const v8bf*)(p + 16);
  v16bf r;
#pragma unroll
  for (int i = 0; i < 8; ++i) { r[i] = lo[i]; r[i + 8] = hi[i]; }
  return r;
}

template <int NTILES, bool RELU>
__global__ void k_gemm_wmma(const __bf16* __restrict__ act,
                            const __bf16* __restrict__ Wc,
                            const float* __restrict__ bias,
                            float* __restrict__ out, int N) {
  const int Cout = NTILES * 16;
  const int lane = threadIdx.x;         // blockDim.x == 32
  const int mblk = blockIdx.x;          // 16 rows per block, N % 16 == 0
  const int ml = lane & 15;
  const int kb = (lane >> 4) * 8;

  const __bf16* arow = act + (size_t)(mblk * 16 + ml) * KDIM;

  v8f acc[NTILES];
  v8f z = {0.f, 0.f, 0.f, 0.f, 0.f, 0.f, 0.f, 0.f};
#pragma unroll
  for (int t = 0; t < NTILES; ++t) acc[t] = z;

#pragma unroll
  for (int k0 = 0; k0 < KDIM; k0 += 32) {
    v16bf a = load_frag(arow + k0 + kb);
#pragma unroll
    for (int t = 0; t < NTILES; ++t) {
      const __bf16* brow = Wc + (size_t)(t * 16 + ml) * KDIM + k0 + kb;
      v16bf b = load_frag(brow);
      acc[t] = __builtin_amdgcn_wmma_f32_16x16x32_bf16(
          /*neg_a=*/false, a, /*neg_b=*/false, b,
          /*c_mod=*/(short)0, acc[t], /*reuse_a=*/false, /*reuse_b=*/false);
    }
  }

  // Epilogue: C/D layout -> VGPR i holds row (i + 8*(lane>>4)), col (lane&15).
  const int mh = (lane >> 4) * 8;
#pragma unroll
  for (int t = 0; t < NTILES; ++t) {
    int n = t * 16 + ml;
    float bv = bias[n];
#pragma unroll
    for (int i = 0; i < 8; ++i) {
      float v = acc[t][i] + bv;
      if (RELU) v = fmaxf(v, 0.0f);
      out[(size_t)(mblk * 16 + mh + i) * Cout + n] = v;
    }
  }
}

// ------------------------------- LayerNorm ---------------------------------
// One wave per row of 128; float4 per lane; xor-ladder wave reduction (wave32).
__global__ void k_layernorm(float* __restrict__ h,
                            const float* __restrict__ g,
                            const float* __restrict__ b, int N) {
  int lane = threadIdx.x & 31;
  int row = (blockIdx.x * blockDim.x + threadIdx.x) >> 5;
  if (row >= N) return;
  float4 v = ((const float4*)(h + (size_t)row * 128))[lane];
  float s  = v.x + v.y + v.z + v.w;
  float sq = v.x * v.x + v.y * v.y + v.z * v.z + v.w * v.w;
#pragma unroll
  for (int off = 16; off > 0; off >>= 1) {
    s  += __shfl_xor(s,  off, 32);
    sq += __shfl_xor(sq, off, 32);
  }
  float mu  = s * (1.0f / 128.0f);
  float var = sq * (1.0f / 128.0f) - mu * mu;
  float rs  = rsqrtf(var + 1e-5f);
  float4 gv = ((const float4*)g)[lane];
  float4 bv = ((const float4*)b)[lane];
  float4 o;
  o.x = (v.x - mu) * rs * gv.x + bv.x;
  o.y = (v.y - mu) * rs * gv.y + bv.y;
  o.z = (v.z - mu) * rs * gv.z + bv.z;
  o.w = (v.w - mu) * rs * gv.w + bv.w;
  ((float4*)(h + (size_t)row * 128))[lane] = o;
}

// ------------------------------- launcher ----------------------------------

extern "C" void kernel_launch(void* const* d_in, const int* in_sizes, int n_in,
                              void* d_out, int out_size, void* d_ws, size_t ws_size,
                              hipStream_t stream) {
  const float* x    = (const float*)d_in[0];
  const int*   ei   = (const int*)d_in[1];
  const float* Wl1  = (const float*)d_in[2];
  const float* bl1  = (const float*)d_in[3];
  const float* Wr1  = (const float*)d_in[4];
  const float* Wl2  = (const float*)d_in[5];
  const float* bl2  = (const float*)d_in[6];
  const float* Wr2  = (const float*)d_in[7];
  const float* Wl3  = (const float*)d_in[8];
  const float* bl3  = (const float*)d_in[9];
  const float* Wr3  = (const float*)d_in[10];
  const float* g1   = (const float*)d_in[11];
  const float* be1  = (const float*)d_in[12];
  const float* g2   = (const float*)d_in[13];
  const float* be2  = (const float*)d_in[14];

  const int N = in_sizes[0] / 128;       // 50000 (divisible by 16)
  const int E = in_sizes[1] / 2;         // 800000
  const int* src = ei;
  const int* dst = ei + E;

  // workspace carve-up (all offsets 512B-aligned)
  char* ws = (char*)d_ws;
  const size_t szF = (size_t)N * 128 * sizeof(float);          // 25.6 MB
  const size_t szA = (size_t)N * KDIM * sizeof(__bf16);        // 25.6 MB
  float*  F0  = (float*)(ws);
  float*  F1  = (float*)(ws + szF);
  __bf16* ACT = (__bf16*)(ws + 2 * szF);
  float*  INV = (float*)(ws + 2 * szF + szA);
  __bf16* WC  = (__bf16*)(ws + 2 * szF + szA + ((size_t)N * 4 + 511) / 512 * 512);

  const long long nodeFeat = (long long)N * 128;
  dim3 z_grid(2048), b256(256), b128(128), b32(32);
  int eWaveBlocks = (E * 32 + 255) / 256;        // one wave per edge
  int lnBlocks    = (N * 32 + 255) / 256;        // one wave per row
  int mBlocks     = N / 16;                      // 3125

  // --- degree counts -> reciprocal (reused by all 3 layers)
  k_zero_f32<<<z_grid, b256, 0, stream>>>(INV, N);
  k_count<<<(E + 255) / 256, b256, 0, stream>>>(dst, INV, E);
  k_invert<<<(N + 255) / 256, b256, 0, stream>>>(INV, N);

  // ---------------- layer 1: x -> F1 ----------------
  k_zero_f32<<<z_grid, b256, 0, stream>>>(F0, nodeFeat);
  k_aggregate<<<eWaveBlocks, b256, 0, stream>>>(x, src, dst, F0, E);
  k_build_act<<<N, b128, 0, stream>>>(F0, INV, x, ACT, N);
  k_pack_w<<<(128 * KDIM + 255) / 256, b256, 0, stream>>>(Wl1, Wr1, WC, 128);
  k_gemm_wmma<8, true><<<mBlocks, b32, 0, stream>>>(ACT, WC, bl1, F1, N);
  k_layernorm<<<lnBlocks, b256, 0, stream>>>(F1, g1, be1, N);

  // ---------------- layer 2: F1 -> F0 ----------------
  k_zero_f32<<<z_grid, b256, 0, stream>>>(F0, nodeFeat);
  k_aggregate<<<eWaveBlocks, b256, 0, stream>>>(F1, src, dst, F0, E);
  k_build_act<<<N, b128, 0, stream>>>(F0, INV, F1, ACT, N);
  k_pack_w<<<(128 * KDIM + 255) / 256, b256, 0, stream>>>(Wl2, Wr2, WC, 128);
  k_gemm_wmma<8, true><<<mBlocks, b32, 0, stream>>>(ACT, WC, bl2, F0, N);   // sums dead after build_act
  k_layernorm<<<lnBlocks, b256, 0, stream>>>(F0, g2, be2, N);

  // ---------------- layer 3: F0 -> d_out (no relu/LN) ----------------
  k_zero_f32<<<z_grid, b256, 0, stream>>>(F1, nodeFeat);
  k_aggregate<<<eWaveBlocks, b256, 0, stream>>>(F0, src, dst, F1, E);
  k_build_act<<<N, b128, 0, stream>>>(F1, INV, F0, ACT, N);
  k_pack_w<<<(64 * KDIM + 255) / 256, b256, 0, stream>>>(Wl3, Wr3, WC, 64);
  k_gemm_wmma<4, false><<<mBlocks, b32, 0, stream>>>(ACT, WC, bl3, (float*)d_out, N);

  (void)n_in; (void)out_size; (void)ws_size;
}
